// Mask_Temporal_Attention_17145509446377
// MI455X (gfx1250) — matrix operand order repaired and checked by reading, not compiled, not run on hardware
//
#include <hip/hip_runtime.h>
#include <math.h>

// Problem constants (from reference)
#define BB   8
#define NN   307
#define LQ   64      // L
#define DM   64      // D_MODEL
#define HH   4       // heads
#define DK   16      // D_K == D_V

typedef __attribute__((ext_vector_type(16))) _Float16 v16h;
typedef __attribute__((ext_vector_type(8)))  _Float16 v8h;
typedef __attribute__((ext_vector_type(4)))  _Float16 v4h;
typedef __attribute__((ext_vector_type(8)))  float    v8f;
typedef __attribute__((ext_vector_type(4)))  float    v4f;

// ---- WMMA fragment helpers (wave32, v_wmma_f32_16x16x32_f16 layouts) ----
// A: 16x32 f16, lane l (grp=l>>4, m=l&15): halves 0..7 -> k=grp*8+0..7,
// halves 8..15 -> k=16+grp*8+0..7  (two contiguous 8-half runs -> 2x b128).
// klim in {16,32} is chunk-aligned, so zeroing is chunk-granular.
__device__ __forceinline__ v16h frag_a(const _Float16* base, int m0, int ld,
                                       int k0, int klim) {
  const int lane = threadIdx.x & 31;
  const int grp  = lane >> 4;
  const int m    = lane & 15;
  const _Float16* rp = base + (m0 + m) * ld + k0;
  const int kb0 = grp * 8, kb1 = 16 + grp * 8;
  v8h z  = {};
  v8h lo = (kb0 + 8 <= klim) ? *(const v8h*)(rp + kb0) : z;
  v8h hi = (kb1 + 8 <= klim) ? *(const v8h*)(rp + kb1) : z;
  v16h a;
#pragma unroll
  for (int i = 0; i < 8; ++i) { a[i] = lo[i]; a[8 + i] = hi[i]; }
  return a;
}

// B: 32x16 f16, source stored as B^T row-major: src[n][k].
// lane l holds column n=l&15; halves 0..7 -> k=grp*8+i, 8..15 -> k=16+grp*8+i.
__device__ __forceinline__ v16h frag_bt(const _Float16* base, int n0, int ld,
                                        int k0, int klim) {
  const int lane = threadIdx.x & 31;
  const int grp  = lane >> 4;
  const int n    = lane & 15;
  const _Float16* rp = base + (n0 + n) * ld + k0;
  const int kb0 = grp * 8, kb1 = 16 + grp * 8;
  v8h z  = {};
  v8h lo = (kb0 + 8 <= klim) ? *(const v8h*)(rp + kb0) : z;
  v8h hi = (kb1 + 8 <= klim) ? *(const v8h*)(rp + kb1) : z;
  v16h b;
#pragma unroll
  for (int i = 0; i < 8; ++i) { b[i] = lo[i]; b[8 + i] = hi[i]; }
  return b;
}

// D/C 16x16 f32: lane l -> n=l&15, rows m = 8*(l>>4)+r for VGPR r.
__device__ __forceinline__ void store_d_f16(_Float16* dst, int m0, int n0,
                                            int ld, v8f d) {
  const int lane = threadIdx.x & 31;
  const int n    = lane & 15;
  const int mb   = (lane >> 4) * 8;
#pragma unroll
  for (int r = 0; r < 8; ++r)
    dst[(m0 + mb + r) * ld + n0 + n] = (_Float16)d[r];
}
// Transposed store: dst[(n0+n)][m0+mb..+7] -> 8 contiguous halves = 1x b128.
__device__ __forceinline__ void store_d_f16_T(_Float16* dst, int m0, int n0,
                                              int ld, v8f d) {
  const int lane = threadIdx.x & 31;
  const int n    = lane & 15;
  const int mb   = (lane >> 4) * 8;
  v8h h;
#pragma unroll
  for (int r = 0; r < 8; ++r) h[r] = (_Float16)d[r];
  *(v8h*)(dst + (n0 + n) * ld + m0 + mb) = h;
}
__device__ __forceinline__ void store_d_f32(float* dst, int m0, int n0,
                                            int ld, v8f d) {
  const int lane = threadIdx.x & 31;
  const int n    = lane & 15;
  const int mb   = (lane >> 4) * 8;
#pragma unroll
  for (int r = 0; r < 8; ++r)
    dst[(m0 + mb + r) * ld + n0 + n] = d[r];
}

#define WMMA_F16(a, b, c)                                                     \
  __builtin_amdgcn_wmma_f32_16x16x32_f16(false, (a), false, (b), (short)0,    \
                                         (c), false, false)

// LDS carve (halves unless noted):
//  sX   [3*4096]  input Q,K,V f16, row-major [l][f]      (dead after Phase B)
//  sW   [4*4096]  W_Q^T,W_K^T,W_V^T,W_fc^T f16           (first 3 dead after B)
//  sP   [3*4096]  Q[l][c], K[l][c], V^T[c][l] f16
//  sA   [4*4096]  attn f16 per head [q][k]
//  sC   [4096]    context f16 [l][c]
//  sO   [4096]    fc output f32                          (dead until Phase F)
//  sM   [64]      mask f32
// Score scratch (f32, Phases C/D only) aliases dead space:
//  heads 0..2 -> bytes [0, 49152) = sX + sWq/k/v ; head 3 -> sO region.
#define SMEM_BYTES ((3+4+3+4)*4096*2 + 4096*2 + 4096*4 + 64*4)   // 139520

__global__ void __launch_bounds__(256)
mta_fused_kernel(const float* __restrict__ gQ, const float* __restrict__ gK,
                 const float* __restrict__ gV, const float* __restrict__ gMask,
                 const float* __restrict__ gRes,
                 const float* __restrict__ WQ, const float* __restrict__ WK,
                 const float* __restrict__ WV, const float* __restrict__ Wfc,
                 const float* __restrict__ lnS, const float* __restrict__ lnB,
                 float* __restrict__ outY, float* __restrict__ outS) {
  extern __shared__ char smem[];
  _Float16* sX = (_Float16*)smem;          // 3*4096
  _Float16* sW = sX + 3 * 4096;            // 4*4096 (transposed weights)
  _Float16* sP = sW + 4 * 4096;            // 3*4096
  _Float16* sA = sP + 3 * 4096;            // 4*4096
  _Float16* sC = sA + 4 * 4096;            // 4096
  float*    sO = (float*)(sC + 4096);      // 4096 f32
  float*    sM = sO + 4096;                // 64 f32
  // f32 score scratch per head (aliases dead regions, see carve comment)
  float* sS0 = (float*)smem;               // heads 0..2: h*4096 floats
  float* sS3 = sO;                         // head 3

  const int tid  = threadIdx.x;
  const int wave = tid >> 5;
  const int blk  = blockIdx.x;             // b*N + n

  const size_t xoff = (size_t)blk * (LQ * DM);
  const float* bq  = gQ + xoff;
  const float* bk  = gK + xoff;
  const float* bv  = gV + xoff;
  const float* res = gRes + (size_t)blk * (HH * LQ * LQ);
  float*       sc  = outS + (size_t)blk * (HH * LQ * LQ);
  float*       yo  = outY + xoff;

  // ---- Phase A: stage inputs (f32->f16, vectorized) + transposed weights ----
  for (int i4 = tid; i4 < 1024; i4 += 256) {          // 4096 halves / 4
    v4f q = ((const v4f*)bq)[i4];
    v4f k = ((const v4f*)bk)[i4];
    v4f v = ((const v4f*)bv)[i4];
    v4h hq, hk, hv;
#pragma unroll
    for (int j = 0; j < 4; ++j) {
      hq[j] = (_Float16)q[j]; hk[j] = (_Float16)k[j]; hv[j] = (_Float16)v[j];
    }
    ((v4h*)sX)[i4]        = hq;
    ((v4h*)sX)[1024 + i4] = hk;
    ((v4h*)sX)[2048 + i4] = hv;
  }
  // Weights staged transposed: sW[e*64+f] = W[f*64+e]. Weights are 16KB and
  // shared by all 2456 blocks -> the strided global gather stays L2-resident.
  for (int i4 = tid; i4 < 1024; i4 += 256) {
    const int i = i4 * 4;
    const int e = i >> 6, f = i & 63;
    v4h wq, wk, wv, wf;
#pragma unroll
    for (int j = 0; j < 4; ++j) {
      wq[j] = (_Float16)WQ [(f + j) * 64 + e];
      wk[j] = (_Float16)WK [(f + j) * 64 + e];
      wv[j] = (_Float16)WV [(f + j) * 64 + e];
      wf[j] = (_Float16)Wfc[(f + j) * 64 + e];
    }
    ((v4h*)sW)[i4]        = wq;
    ((v4h*)sW)[1024 + i4] = wk;
    ((v4h*)sW)[2048 + i4] = wv;
    ((v4h*)sW)[3072 + i4] = wf;
  }
  if (tid < LQ) sM[tid] = gMask[(size_t)blk * LQ + tid];
  __syncthreads();

  // ---- Phase B: Q/K/V projections, 3 x (64x64x64), 48 tile-jobs ----
#pragma unroll
  for (int j = wave; j < 48; j += 8) {
    const int mat = j >> 4;
    const int t   = j & 15;
    const int m0  = (t >> 2) * 16, n0 = (t & 3) * 16;
    v8f acc = {};
#pragma unroll
    for (int k0 = 0; k0 < 64; k0 += 32) {
      v16h a = frag_a (sX + mat * 4096, m0, 64, k0, 32);
      v16h b = frag_bt(sW + mat * 4096, n0, 64, k0, 32);  // W^T[e][f]
      acc = WMMA_F16(a, b, acc);
    }
    if (mat == 2)
      store_d_f16_T(sP + 8192, m0, n0, 64, acc);  // V stored transposed [c][l]
    else
      store_d_f16(sP + mat * 4096, m0, n0, 64, acc);
  }
  __syncthreads();

  const _Float16* sQp = sP;            // [q][c]
  const _Float16* sKp = sP + 4096;     // [k][c]
  const _Float16* sVt = sP + 8192;     // [c][k]  (transposed)

  // ---- Phase C: scores = Q·K^T/4 + res, masked ----
  // Dual write: global (tuple output) + LDS score scratch for the softmax.
#pragma unroll
  for (int j = wave; j < 64; j += 8) {
    const int h  = j >> 4;
    const int t  = j & 15;
    const int q0 = (t >> 2) * 16, kk0 = (t & 3) * 16;
    v16h a = frag_a (sQp, q0,  64, h * 16, DK);   // d>=16 halves -> 0
    v16h b = frag_bt(sKp, kk0, 64, h * 16, DK);   // K rows are B^T source
    v8f acc = {};
    acc = WMMA_F16(a, b, acc);
    float* sSh = (h < 3) ? (sS0 + h * 4096) : sS3;
    const int lane = tid & 31;
    const int n    = lane & 15;
    const int mb   = (lane >> 4) * 8;
#pragma unroll
    for (int r = 0; r < 8; ++r) {
      const int q = q0 + mb + r;
      const int k = kk0 + n;
      float s = acc[r] * 0.25f + res[h * 4096 + q * 64 + k];
      if (sM[q] > 0.5f) s = -1.0e9f;               // mask broadcasts over k
      sc[h * 4096 + q * 64 + k]  = s;              // scores output (global)
      sSh[q * 64 + k]            = s;              // softmax scratch (LDS)
    }
  }
  __syncthreads();

  // ---- Phase D: softmax over the *query* axis (reference axis=3) ----
  // thread t -> (h = t/64, k = t%64); pure LDS column reads, lane<->k maps
  // to distinct banks at the 256B row stride -> conflict-free.
  {
    const int h = tid >> 6, k = tid & 63;
    const float* col = ((h < 3) ? (sS0 + h * 4096) : sS3) + k;
    float mx = -3.4e38f;
    for (int q = 0; q < 64; ++q) mx = fmaxf(mx, col[q * 64]);
    float sum = 0.f;
    for (int q = 0; q < 64; ++q) sum += __expf(col[q * 64] - mx);
    const float inv = 1.f / sum;
    for (int q = 0; q < 64; ++q)
      sA[h * 4096 + q * 64 + k] = (_Float16)(__expf(col[q * 64] - mx) * inv);
  }
  __syncthreads();

  // ---- Phase E: context = attn @ V_h  (k=64 contraction) ----
#pragma unroll
  for (int j = wave; j < 16; j += 8) {
    const int h  = j >> 2;
    const int q0 = (j & 3) * 16;
    v8f acc = {};
#pragma unroll
    for (int k0 = 0; k0 < 64; k0 += 32) {
      v16h a = frag_a (sA + h * 4096, q0, 64, k0, 32);
      v16h b = frag_bt(sVt, h * 16, 64, k0, 32);   // V^T[c][k]
      acc = WMMA_F16(a, b, acc);
    }
    store_d_f16(sC, q0, h * 16, 64, acc);          // context[l][h*16+d]
  }
  __syncthreads();

  // ---- Phase F: output = context @ W_fc ----
#pragma unroll
  for (int j = wave; j < 16; j += 8) {
    const int m0 = (j >> 2) * 16, n0 = (j & 3) * 16;
    v8f acc = {};
#pragma unroll
    for (int k0 = 0; k0 < 64; k0 += 32) {
      v16h a = frag_a (sC, m0, 64, k0, 32);
      v16h b = frag_bt(sW + 12288, n0, 64, k0, 32); // W_fc^T[e][f]
      acc = WMMA_F16(a, b, acc);
    }
    store_d_f32(sO, m0, n0, 64, acc);               // overwrites head-3 scratch
  }
  __syncthreads();

  // ---- Phase G: residual + layernorm (row per thread, rows = 64) ----
  if (tid < LQ) {
    const int row = tid;
    float mu = 0.f;
    for (int f = 0; f < DM; ++f) {
      float v = sO[row * 64 + f] + bq[row * 64 + f];
      sO[row * 64 + f] = v;
      mu += v;
    }
    mu *= (1.f / 64.f);
    float var = 0.f;
    for (int f = 0; f < DM; ++f) {
      float d = sO[row * 64 + f] - mu;
      var += d * d;
    }
    var *= (1.f / 64.f);
    const float rstd = rsqrtf(var + 1e-5f);
    for (int f = 0; f < DM; ++f)
      yo[row * 64 + f] = (sO[row * 64 + f] - mu) * rstd * lnS[f] + lnB[f];
  }
}

extern "C" void kernel_launch(void* const* d_in, const int* in_sizes, int n_in,
                              void* d_out, int out_size, void* d_ws,
                              size_t ws_size, hipStream_t stream) {
  (void)in_sizes; (void)n_in; (void)d_ws; (void)ws_size; (void)out_size;
  const float* gQ    = (const float*)d_in[0];
  const float* gK    = (const float*)d_in[1];
  const float* gV    = (const float*)d_in[2];
  const float* gMask = (const float*)d_in[3];
  const float* gRes  = (const float*)d_in[4];
  const float* WQ    = (const float*)d_in[5];
  const float* WK    = (const float*)d_in[6];
  const float* WV    = (const float*)d_in[7];
  const float* Wfc   = (const float*)d_in[8];
  const float* lnS   = (const float*)d_in[9];
  const float* lnB   = (const float*)d_in[10];

  float* outY = (float*)d_out;                                   // (B,N,L,DM)
  float* outS = outY + (size_t)BB * NN * LQ * DM;                // (B,N,H,L,L)

  mta_fused_kernel<<<BB * NN, 256, SMEM_BYTES, stream>>>(
      gQ, gK, gV, gMask, gRes, WQ, WK, WV, Wfc, lnS, lnB, outY, outS);
}
